// BayesLinear_18494129176930
// MI455X (gfx1250) — matrix operand order repaired and checked by reading, not compiled, not run on hardware
//
#include <hip/hip_runtime.h>

// ---------------------------------------------------------------------------
// BayesLinear: out = X @ (mu + softplus(rho)*eps)^T + (bmu + softplus(brho)*beps)
//   X: [8192, 4096] f32, W params: [4096, 4096] f32, out: [8192, 4096] f32
// MI455X strategy:
//   1) sample W -> bf16, convert X -> bf16, sample bias -> f32 (workspace)
//   2) WMMA bf16 GEMM, 128x256 block tile staged through LDS with
//      double-buffered GLOBAL_LOAD_ASYNC_TO_LDS_B128 (ASYNCcnt) copies,
//      64x64 wave tile = 4x4 x v_wmma_f32_16x16x32_bf16, bias fused.
// ---------------------------------------------------------------------------

#define TOKENS 8192
#define D_IN   4096
#define D_OUT  4096

// LDS tile geometry: K-step 32 (64B rows), padded to 80B to spread banks
#define A_STRIDE 80                 // bytes per A row (128 rows)
#define B_STRIDE 80                 // bytes per B row (256 rows)
#define A_TILE   (128 * A_STRIDE)   // 10240 B
#define B_TILE   (256 * B_STRIDE)   // 20480 B

typedef __attribute__((ext_vector_type(16))) __bf16 v16bf;
typedef __attribute__((ext_vector_type(8)))  __bf16 v8bf;
typedef __attribute__((ext_vector_type(4)))  __bf16 v4bf;
typedef __attribute__((ext_vector_type(8)))  float  v8f;

__device__ __forceinline__ float softplus_f(float r) {
    return fmaxf(r, 0.0f) + log1pf(__expf(-fabsf(r)));
}

// ---- elementwise: W = mu + softplus(rho)*eps, cast to bf16 ------------------
__global__ void sample_weight_bf16(const float* __restrict__ mu,
                                   const float* __restrict__ rho,
                                   const float* __restrict__ eps,
                                   __bf16* __restrict__ wout, int n4) {
    int idx = blockIdx.x * blockDim.x + threadIdx.x;
    if (idx >= n4) return;
    float4 m = ((const float4*)mu)[idx];
    float4 r = ((const float4*)rho)[idx];
    float4 e = ((const float4*)eps)[idx];
    v4bf o;
    o[0] = (__bf16)(m.x + softplus_f(r.x) * e.x);
    o[1] = (__bf16)(m.y + softplus_f(r.y) * e.y);
    o[2] = (__bf16)(m.z + softplus_f(r.z) * e.z);
    o[3] = (__bf16)(m.w + softplus_f(r.w) * e.w);
    ((v4bf*)wout)[idx] = o;
}

// ---- elementwise: X f32 -> bf16 --------------------------------------------
__global__ void convert_x_bf16(const float* __restrict__ x,
                               __bf16* __restrict__ xout, int n4) {
    int idx = blockIdx.x * blockDim.x + threadIdx.x;
    if (idx >= n4) return;
    float4 v = ((const float4*)x)[idx];
    v4bf o;
    o[0] = (__bf16)v.x; o[1] = (__bf16)v.y; o[2] = (__bf16)v.z; o[3] = (__bf16)v.w;
    ((v4bf*)xout)[idx] = o;
}

// ---- bias = bmu + softplus(brho)*beps --------------------------------------
__global__ void sample_bias_f32(const float* __restrict__ bmu,
                                const float* __restrict__ brho,
                                const float* __restrict__ beps,
                                float* __restrict__ bout, int n) {
    int idx = blockIdx.x * blockDim.x + threadIdx.x;
    if (idx >= n) return;
    bout[idx] = bmu[idx] + softplus_f(brho[idx]) * beps[idx];
}

// ---- CDNA5 async copy: 16B global -> LDS, tracked by ASYNCcnt --------------
__device__ __forceinline__ void async_copy16(unsigned lds_addr, const void* gaddr) {
    // lds_addr = low 32 bits of generic pointer
    asm volatile("global_load_async_to_lds_b128 %0, %1, off"
                 :: "v"(lds_addr), "v"(gaddr)
                 : "memory");
}
#define S_WAIT_ASYNCCNT(N) asm volatile("s_wait_asynccnt " #N ::: "memory")

// ---- LDS fragment loaders (per CDNA5 WMMA VGPR layouts) --------------------
// A 16-bit 16x32: lane L holds row (L&15); bytes [kb..kb+15] and [kb+32..kb+47]
// of the 64B row, kb = (L>>4)*16.
__device__ __forceinline__ v16bf fragA_lds(const char* As, int row, int lane) {
    const char* p = As + row * A_STRIDE + ((lane >> 4) << 4);
    v8bf lo = *(const v8bf*)p;
    v8bf hi = *(const v8bf*)(p + 32);
    v16bf f;
#pragma unroll
    for (int i = 0; i < 8; ++i) { f[i] = lo[i]; f[i + 8] = hi[i]; }
    return f;
}
// B 32x16: lane L holds column (L&15); bytes [kb..kb+15] and [kb+32..kb+47], kb = (L>>4)*16.
__device__ __forceinline__ v16bf fragB_lds(const char* Bs, int row, int lane) {
    const char* p = Bs + row * B_STRIDE + ((lane >> 4) << 4);
    v8bf lo = *(const v8bf*)p;
    v8bf hi = *(const v8bf*)(p + 32);
    v16bf f;
#pragma unroll
    for (int i = 0; i < 8; ++i) { f[i] = lo[i]; f[i + 8] = hi[i]; }
    return f;
}

// ---- WMMA GEMM with double-buffered async LDS staging ----------------------
__global__ __launch_bounds__(256)
void bayes_gemm_wmma(const __bf16* __restrict__ Xb,
                     const __bf16* __restrict__ Wb,
                     const float*  __restrict__ bias,
                     float* __restrict__ out) {
    __shared__ __align__(32) char smem[2 * A_TILE + 2 * B_TILE];  // 60 KB

    const int tid  = threadIdx.x;
    const int lane = tid & 31;
    const int wave = tid >> 5;
    const int wm = wave >> 2;            // 0..1
    const int wn = wave & 3;             // 0..3
    const int mBlk = blockIdx.y * 128;
    const int nBlk = blockIdx.x * 256;
    const int m0 = wm * 64;              // wave tile origin within block tile
    const int n0 = wn * 64;

    // Issue async copies for one K-tile into buffer `buf` (6 instructions/wave):
    // A: 512 x 16B chunks (2/thread), B: 1024 x 16B chunks (4/thread).
    // Consecutive lanes take consecutive 16B chunks -> fully coalesced.
#define ISSUE_TILE(buf, kk) do {                                                 \
        char* As_ = smem + (buf) * A_TILE;                                       \
        char* Bs_ = smem + 2 * A_TILE + (buf) * B_TILE;                          \
        _Pragma("unroll")                                                        \
        for (int q = 0; q < 2; ++q) {                                            \
            int cid = tid + q * 256;                                             \
            int row = cid >> 2, col = cid & 3;                                   \
            const __bf16* g = Xb + (size_t)(mBlk + row) * D_IN + (kk) + col * 8; \
            async_copy16((unsigned)(size_t)(As_ + row * A_STRIDE + col * 16), g);\
        }                                                                        \
        _Pragma("unroll")                                                        \
        for (int q = 0; q < 4; ++q) {                                            \
            int cid = tid + q * 256;                                             \
            int row = cid >> 2, col = cid & 3;                                   \
            const __bf16* g = Wb + (size_t)(nBlk + row) * D_IN + (kk) + col * 8; \
            async_copy16((unsigned)(size_t)(Bs_ + row * B_STRIDE + col * 16), g);\
        }                                                                        \
    } while (0)

    v8f acc[4][4];
#pragma unroll
    for (int i = 0; i < 4; ++i)
#pragma unroll
        for (int j = 0; j < 4; ++j)
            acc[i][j] = (v8f){0.f, 0.f, 0.f, 0.f, 0.f, 0.f, 0.f, 0.f};

    ISSUE_TILE(0, 0);                                  // prologue fill

    const int KSTEPS = D_IN / 32;                      // 128
    for (int it = 0; it < KSTEPS; ++it) {
        // Prefetch next tile into the other buffer (WAR-safe: trailing barrier
        // of iteration it-1 guarantees everyone finished reading it).
        int kn = (it + 1 < KSTEPS) ? (it + 1) * 32 : 0;  // dummy wrap keeps count uniform
        ISSUE_TILE((it + 1) & 1, kn);

        S_WAIT_ASYNCCNT(6);        // the 6 older copies (tile `it`) have landed
        __syncthreads();           // publish LDS writes to all waves

        const char* As = smem + (it & 1) * A_TILE;
        const char* Bs = smem + 2 * A_TILE + (it & 1) * B_TILE;

        v16bf a[4], b[4];
#pragma unroll
        for (int i = 0; i < 4; ++i)
            a[i] = fragA_lds(As, m0 + 16 * i + (lane & 15), lane);
#pragma unroll
        for (int j = 0; j < 4; ++j)
            b[j] = fragB_lds(Bs, n0 + 16 * j + (lane & 15), lane);

#pragma unroll
        for (int i = 0; i < 4; ++i)
#pragma unroll
            for (int j = 0; j < 4; ++j)
                acc[i][j] = __builtin_amdgcn_wmma_f32_16x16x32_bf16(
                    false, a[i], false, b[j], (short)0, acc[i][j], false, false);

        __syncthreads();           // everyone done reading this buffer
    }
#undef ISSUE_TILE

    // Epilogue. C/D layout: lane L, VGPR r -> M = 8*(L>>4)+r, N = (L&15).
#pragma unroll
    for (int j = 0; j < 4; ++j) {
        int n = nBlk + n0 + 16 * j + (lane & 15);
        float bv = bias[n];
#pragma unroll
        for (int i = 0; i < 4; ++i) {
            int mbase = mBlk + m0 + 16 * i + ((lane >> 4) << 3);
            float* op = out + (size_t)mbase * D_OUT + n;
#pragma unroll
            for (int r = 0; r < 8; ++r)
                op[(size_t)r * D_OUT] = acc[i][j][r] + bv;
        }
    }
}

extern "C" void kernel_launch(void* const* d_in, const int* in_sizes, int n_in,
                              void* d_out, int out_size, void* d_ws, size_t ws_size,
                              hipStream_t stream) {
    const float* x    = (const float*)d_in[0];
    const float* wmu  = (const float*)d_in[1];
    const float* wrho = (const float*)d_in[2];
    const float* bmu  = (const float*)d_in[3];
    const float* brho = (const float*)d_in[4];
    const float* ew   = (const float*)d_in[5];
    const float* eb   = (const float*)d_in[6];
    float* out = (float*)d_out;

    char* ws = (char*)d_ws;
    __bf16* Xb   = (__bf16*)ws;                                        // 64 MB
    __bf16* Wb   = (__bf16*)(ws + (size_t)TOKENS * D_IN * 2);          // 32 MB
    float*  bias = (float*)(ws + (size_t)TOKENS * D_IN * 2
                               + (size_t)D_OUT * D_IN * 2);            // 16 KB

    {
        int n4 = (D_OUT * D_IN) / 4;
        sample_weight_bf16<<<n4 / 256, 256, 0, stream>>>(wmu, wrho, ew, Wb, n4);
    }
    {
        int n4 = (TOKENS * D_IN) / 4;
        convert_x_bf16<<<n4 / 256, 256, 0, stream>>>(x, Xb, n4);
    }
    {
        sample_bias_f32<<<D_OUT / 256, 256, 0, stream>>>(bmu, brho, eb, bias, D_OUT);
    }
    {
        dim3 grid(D_OUT / 256, TOKENS / 128);
        bayes_gemm_wmma<<<grid, 256, 0, stream>>>(Xb, Wb, bias, out);
    }
}
